// Trans2GraphConv_68831145886206
// MI455X (gfx1250) — compile-verified, matrix operand order checked by reading, not run on hardware
//
#include <hip/hip_runtime.h>
#include <math.h>

// ---------------- problem constants ----------------
#define NN    32768          // nodes
#define EE    131072         // edges
#define IND   256            // in_dim
#define OUTD  256            // out_dim
#define TDIM  32             // time_dim
#define NH    4              // heads
#define NT    19             // node types
#define PROJD 192            // k/q/v proj dim
#define KQVC  576            // 3*PROJD fused columns
#define MAXPAD   (NN + NT * 16)   // per-type padding to 16-row tiles
#define MAXTILES (MAXPAD / 16)
#define SQRT_DK  7.48331477354788277f   // sqrt(56)
#define LN_EPS   1e-5f

// ---------------- WMMA fragment types (CDNA5, wave32) ----------------
typedef __attribute__((ext_vector_type(16))) __bf16 v16bf;
typedef __attribute__((ext_vector_type(8)))  __bf16 v8bf;
typedef __attribute__((ext_vector_type(8)))  float  v8f;

union FragU { v16bf v; v8bf h[2]; };
union BfU   { unsigned short u; __bf16 b; };

__device__ __forceinline__ __bf16 f2bf(float f) {
  unsigned u = __float_as_uint(f);
  u = u + 0x7FFFu + ((u >> 16) & 1u);          // round-to-nearest-even
  BfU r; r.u = (unsigned short)(u >> 16);
  return r.b;
}

// order-preserving float<->uint key for atomicMax-based segment max
__device__ __forceinline__ unsigned fkey(float f) {
  unsigned u = __float_as_uint(f);
  return (u & 0x80000000u) ? ~u : (u | 0x80000000u);
}
__device__ __forceinline__ float fkeydec(unsigned k) {
  unsigned u = (k & 0x80000000u) ? (k & 0x7FFFFFFFu) : ~k;
  return __uint_as_float(u);
}

// ---------------- phase 0: type bucketing ----------------
__global__ void count_types_kernel(const int* __restrict__ ntype, int* __restrict__ cnt) {
  int n = blockIdx.x * blockDim.x + threadIdx.x;
  if (n >= NN) return;
  atomicAdd(&cnt[ntype[n]], 1);
}

__global__ void scan_kernel(const int* __restrict__ cnt, int* __restrict__ offs,
                            int* __restrict__ tileType) {
  if (blockIdx.x != 0 || threadIdx.x != 0) return;
  int off = 0;
  for (int t = 0; t < NT; ++t) {
    offs[t] = off;
    int padded = (cnt[t] + 15) & ~15;
    int tbase  = off >> 4;
    int tiles  = padded >> 4;
    for (int i = 0; i < tiles; ++i) tileType[tbase + i] = t;
    off += padded;
  }
  offs[NT] = off;
}

__global__ void build_perm_kernel(const int* __restrict__ ntype, const int* __restrict__ offs,
                                  int* __restrict__ cur, int* __restrict__ perm) {
  int n = blockIdx.x * blockDim.x + threadIdx.x;
  if (n >= NN) return;
  int t = ntype[n];
  int pos = offs[t] + atomicAdd(&cur[t], 1);
  perm[pos] = n;
}

// ---------------- phase 1: bf16 conversions / weight packing ----------------
__global__ void cvt_x_kernel(const float* __restrict__ x, __bf16* __restrict__ xb) {
  size_t i = (size_t)blockIdx.x * blockDim.x + threadIdx.x;
  if (i >= (size_t)NN * IND) return;
  xb[i] = f2bf(x[i]);
}

// wt[t][col][k] : col<192 -> Wk, <384 -> Wq, else Wv  (transposed so B loads are contiguous in k)
__global__ void cvt_wkqv_kernel(const float* __restrict__ Wk, const float* __restrict__ Wq,
                                const float* __restrict__ Wv, __bf16* __restrict__ wt) {
  size_t i = (size_t)blockIdx.x * blockDim.x + threadIdx.x;
  if (i >= (size_t)NT * KQVC * IND) return;
  int k   = (int)(i % IND);
  int c   = (int)((i / IND) % KQVC);
  int t   = (int)(i / ((size_t)IND * KQVC));
  float v;
  if (c < PROJD)          v = Wk[((size_t)t * IND + k) * PROJD + c];
  else if (c < 2 * PROJD) v = Wq[((size_t)t * IND + k) * PROJD + (c - PROJD)];
  else                    v = Wv[((size_t)t * IND + k) * PROJD + (c - 2 * PROJD)];
  wt[i] = f2bf(v);
}

__global__ void build_bkqv_kernel(const float* __restrict__ bk, const float* __restrict__ bq,
                                  const float* __restrict__ bv, float* __restrict__ bkqv) {
  int i = blockIdx.x * blockDim.x + threadIdx.x;
  if (i >= NT * KQVC) return;
  int c = i % KQVC, t = i / KQVC;
  float v;
  if (c < PROJD)          v = bk[t * PROJD + c];
  else if (c < 2 * PROJD) v = bq[t * PROJD + (c - PROJD)];
  else                    v = bv[t * PROJD + (c - 2 * PROJD)];
  bkqv[i] = v;
}

// wast[t][col][k] = Wa[t][k][col] + Wa[t][256+k][col]   (h = concat(agg,agg) folded)
__global__ void cvt_wast_kernel(const float* __restrict__ Wa, __bf16* __restrict__ wast) {
  size_t i = (size_t)blockIdx.x * blockDim.x + threadIdx.x;
  if (i >= (size_t)NT * OUTD * IND) return;
  int k = (int)(i % IND);
  int c = (int)((i / IND) % OUTD);
  int t = (int)(i / ((size_t)IND * OUTD));
  float v = Wa[((size_t)t * 512 + k) * IND + c] + Wa[((size_t)t * 512 + 256 + k) * IND + c];
  wast[i] = f2bf(v);
}

// ---------------- phase 2: grouped GEMMs via v_wmma_f32_16x16x32_bf16 ----------------
// One wave computes a 16x64 output slab: 1 A fragment amortized over 4 N-tiles.
// Fragments are double-buffered: loads for K-step s+1 are issued before the
// 4 WMMAs of K-step s, so waits are partial instead of full drains.

__device__ __forceinline__ void load_frag(FragU& f, const __bf16* p, int kk, int half) {
  f.h[0] = *(const v8bf*)(p + kk + half * 8);
  f.h[1] = *(const v8bf*)(p + kk + 16 + half * 8);
}

__device__ __forceinline__ void gemm_slab(const __bf16* arow, const __bf16* brow0,
                                          int half, v8f& acc0, v8f& acc1,
                                          v8f& acc2, v8f& acc3) {
  const __bf16* brow1 = brow0 + (size_t)16 * IND;
  const __bf16* brow2 = brow0 + (size_t)32 * IND;
  const __bf16* brow3 = brow0 + (size_t)48 * IND;

  FragU a[2], b0[2], b1[2], b2[2], b3[2];
  load_frag(a[0],  arow,  0, half);
  load_frag(b0[0], brow0, 0, half);
  load_frag(b1[0], brow1, 0, half);
  load_frag(b2[0], brow2, 0, half);
  load_frag(b3[0], brow3, 0, half);

#pragma unroll
  for (int s = 0; s < IND / 32; ++s) {
    const int cur = s & 1;
    const int nxt = cur ^ 1;
    if (s < IND / 32 - 1) {
      const int kn = (s + 1) * 32;
      load_frag(a[nxt],  arow,  kn, half);
      load_frag(b0[nxt], brow0, kn, half);
      load_frag(b1[nxt], brow1, kn, half);
      load_frag(b2[nxt], brow2, kn, half);
      load_frag(b3[nxt], brow3, kn, half);
    }
    acc0 = __builtin_amdgcn_wmma_f32_16x16x32_bf16(false, a[cur].v, false, b0[cur].v, (short)0, acc0, false, false);
    acc1 = __builtin_amdgcn_wmma_f32_16x16x32_bf16(false, a[cur].v, false, b1[cur].v, (short)0, acc1, false, false);
    acc2 = __builtin_amdgcn_wmma_f32_16x16x32_bf16(false, a[cur].v, false, b2[cur].v, (short)0, acc2, false, false);
    acc3 = __builtin_amdgcn_wmma_f32_16x16x32_bf16(false, a[cur].v, false, b3[cur].v, (short)0, acc3, false, false);
  }
}

__global__ __launch_bounds__(32) void gemm_kqv_kernel(
    const __bf16* __restrict__ xb, const __bf16* __restrict__ wt,
    const float* __restrict__ bkqv, const int* __restrict__ perm,
    const int* __restrict__ tileType,
    float* __restrict__ Kout, float* __restrict__ Qout, float* __restrict__ Vout) {
  const int tm = blockIdx.x;
  const int tt = tileType[tm];
  if (tt < 0) return;                       // past the padded row range (uniform per wave)
  const int lane = threadIdx.x;
  const int half = lane >> 4;
  const int idx  = lane & 15;
  const int colBase = blockIdx.y * 64;      // 4 tiles of 16 cols; 0..512

  const int anode = perm[tm * 16 + idx];
  const __bf16* arow  = xb + (size_t)(anode >= 0 ? anode : 0) * IND;
  const __bf16* brow0 = wt + ((size_t)tt * KQVC + colBase + idx) * IND;

  v8f acc0 = {}, acc1 = {}, acc2 = {}, acc3 = {};
  gemm_slab(arow, brow0, half, acc0, acc1, acc2, acc3);

  // C/D layout: VGPR i holds M=i (lanes 0-15) / M=8+i (lanes 16-31), N=lane&15
  int nodes[8];
#pragma unroll
  for (int i = 0; i < 8; ++i) nodes[i] = perm[tm * 16 + i + 8 * half];

#pragma unroll
  for (int j = 0; j < 4; ++j) {
    const v8f acc = (j == 0) ? acc0 : (j == 1) ? acc1 : (j == 2) ? acc2 : acc3;
    const int col = colBase + j * 16 + idx;
    const float bias = bkqv[tt * KQVC + col];
#pragma unroll
    for (int i = 0; i < 8; ++i) {
      const int node = nodes[i];
      if (node < 0) continue;               // padding row
      const float val = acc[i] + bias;
      if (col < PROJD)          Kout[(size_t)node * PROJD + col] = val;
      else if (col < 2 * PROJD) Qout[(size_t)node * PROJD + (col - PROJD)] = val;
      else                      Vout[(size_t)node * PROJD + (col - 2 * PROJD)] = val;
    }
  }
}

__global__ __launch_bounds__(32) void gemm_out_kernel(
    const __bf16* __restrict__ aggb, const __bf16* __restrict__ wast,
    const float* __restrict__ ba, const int* __restrict__ perm,
    const int* __restrict__ tileType, float* __restrict__ o) {
  const int tm = blockIdx.x;
  const int tt = tileType[tm];
  if (tt < 0) return;
  const int lane = threadIdx.x;
  const int half = lane >> 4;
  const int idx  = lane & 15;
  const int colBase = blockIdx.y * 64;      // 0..192

  const int anode = perm[tm * 16 + idx];
  const __bf16* arow  = aggb + (size_t)(anode >= 0 ? anode : 0) * OUTD;
  const __bf16* brow0 = wast + ((size_t)tt * OUTD + colBase + idx) * IND;

  v8f acc0 = {}, acc1 = {}, acc2 = {}, acc3 = {};
  gemm_slab(arow, brow0, half, acc0, acc1, acc2, acc3);

  int nodes[8];
#pragma unroll
  for (int i = 0; i < 8; ++i) nodes[i] = perm[tm * 16 + i + 8 * half];

#pragma unroll
  for (int j = 0; j < 4; ++j) {
    const v8f acc = (j == 0) ? acc0 : (j == 1) ? acc1 : (j == 2) ? acc2 : acc3;
    const int col = colBase + j * 16 + idx;
    const float bias = ba[tt * IND + col];
#pragma unroll
    for (int i = 0; i < 8; ++i) {
      const int node = nodes[i];
      if (node < 0) continue;
      o[(size_t)node * IND + col] = acc[i] + bias;
    }
  }
}

// ---------------- phase 3: edge pipeline ----------------
__global__ void time_emb_kernel(const float* __restrict__ eattr,
                                const float* __restrict__ wtime, const float* __restrict__ btime,
                                float* __restrict__ e0, float* __restrict__ e1,
                                float* __restrict__ e2) {
  size_t i = (size_t)blockIdx.x * blockDim.x + threadIdx.x;
  if (i >= (size_t)3 * EE * TDIM) return;
  int j     = (int)(i & (TDIM - 1));
  int e     = (int)((i >> 5) % EE);
  int which = (int)(i / ((size_t)EE * TDIM));
  float t = eattr[(size_t)e * 5 + which];
  float v = cosf(logf(t + 1.0f) * wtime[j] + btime[j]);
  float* dst = (which == 0) ? e0 : (which == 1) ? e1 : e2;
  dst[(size_t)e * TDIM + j] = v;
}

__global__ void edge_att_kernel(const int* __restrict__ eidx,
                                const float* __restrict__ K, const float* __restrict__ Q,
                                const float* __restrict__ e1, const float* __restrict__ e2,
                                float* __restrict__ attF, float* __restrict__ attB,
                                unsigned* __restrict__ mF, unsigned* __restrict__ mB) {
  int e = blockIdx.x * blockDim.x + threadIdx.x;
  if (e >= EE) return;
  const int row = eidx[e], col = eidx[EE + e];
  const float* Kr = K + (size_t)row * PROJD;
  const float* Qc = Q + (size_t)col * PROJD;
  const float* Kc = K + (size_t)col * PROJD;
  const float* Qr = Q + (size_t)row * PROJD;
  const float* E1 = e1 + (size_t)e * TDIM;
  const float* E2 = e2 + (size_t)e * TDIM;

  float aF[NH] = {0.f, 0.f, 0.f, 0.f};
  float aB[NH] = {0.f, 0.f, 0.f, 0.f};
  for (int d = 0; d < 56; ++d) {
    float qsF = 0.f, qsB = 0.f;
#pragma unroll
    for (int h = 0; h < NH; ++h) {
      int ix = h * 56 + d;
      qsF += (ix < PROJD) ? Qc[ix] : E2[ix - PROJD];   // forward q = [Q[col], e2]
      qsB += (ix < PROJD) ? Qr[ix] : E1[ix - PROJD];   // backward q = [Q[row], e1]
    }
#pragma unroll
    for (int h = 0; h < NH; ++h) {
      int ix = h * 56 + d;
      float kF = (ix < PROJD) ? Kr[ix] : E1[ix - PROJD];   // forward k = [K[row], e1]
      float kB = (ix < PROJD) ? Kc[ix] : E2[ix - PROJD];   // backward k = [K[col], e2]
      aF[h] += kF * qsF;
      aB[h] += kB * qsB;
    }
  }
#pragma unroll
  for (int h = 0; h < NH; ++h) {
    float vF = aF[h] / SQRT_DK;
    float vB = aB[h] / SQRT_DK;
    attF[(size_t)e * NH + h] = vF;
    attB[(size_t)e * NH + h] = vB;
    atomicMax(&mF[(size_t)col * NH + h], fkey(vF));
    atomicMax(&mB[(size_t)row * NH + h], fkey(vB));
  }
}

__global__ void edge_exp_kernel(const int* __restrict__ eidx,
                                float* __restrict__ attF, float* __restrict__ attB,
                                const unsigned* __restrict__ mF, const unsigned* __restrict__ mB,
                                float* __restrict__ sF, float* __restrict__ sB) {
  int e = blockIdx.x * blockDim.x + threadIdx.x;
  if (e >= EE) return;
  const int row = eidx[e], col = eidx[EE + e];
#pragma unroll
  for (int h = 0; h < NH; ++h) {
    float xF = expf(attF[(size_t)e * NH + h] - fkeydec(mF[(size_t)col * NH + h]));
    float xB = expf(attB[(size_t)e * NH + h] - fkeydec(mB[(size_t)row * NH + h]));
    attF[(size_t)e * NH + h] = xF;
    attB[(size_t)e * NH + h] = xB;
    atomicAdd(&sF[(size_t)col * NH + h], xF);
    atomicAdd(&sB[(size_t)row * NH + h], xB);
  }
}

// one thread per (edge, 8-column chunk); 56/112/168/192/224 are all multiples of 8,
// so each chunk is uniform in head and source (V vs time-embedding)
__global__ void edge_scatter_kernel(const int* __restrict__ eidx,
                                    const float* __restrict__ V,
                                    const float* __restrict__ e0, const float* __restrict__ e1,
                                    const float* __restrict__ e2,
                                    const float* __restrict__ attF, const float* __restrict__ attB,
                                    const float* __restrict__ sF, const float* __restrict__ sB,
                                    float* __restrict__ agg) {
  size_t t = (size_t)blockIdx.x * blockDim.x + threadIdx.x;
  if (t >= (size_t)EE * 32) return;
  const int e  = (int)(t >> 5);
  const int c0 = (int)(t & 31) * 8;
  const int row = eidx[e], col = eidx[EE + e];
  float* aggc = agg + (size_t)col * OUTD;
  float* aggr = agg + (size_t)row * OUTD;

  if (c0 >= 224) {
    const float* E0 = e0 + (size_t)e * TDIM + (c0 - 224);
#pragma unroll
    for (int j = 0; j < 8; ++j) {
      float v = E0[j];
      atomicAdd(&aggc[c0 + j], v);   // forward res tail = e0
      atomicAdd(&aggr[c0 + j], v);   // backward res_i tail = e0
    }
  } else {
    const int h = c0 / 56;
    const float wF = attF[(size_t)e * NH + h] / (sF[(size_t)col * NH + h] + 1e-16f);
    const float wB = attB[(size_t)e * NH + h] / (sB[(size_t)row * NH + h] + 1e-16f);
    const float* Vr = V + (size_t)row * PROJD;
    const float* Vc = V + (size_t)col * PROJD;
    const float* E1 = e1 + (size_t)e * TDIM;
    const float* E2 = e2 + (size_t)e * TDIM;
#pragma unroll
    for (int j = 0; j < 8; ++j) {
      int c = c0 + j;
      float vF = (c < PROJD) ? Vr[c] : E1[c - PROJD];   // forward v = [V[row], e1]
      float vB = (c < PROJD) ? Vc[c] : E2[c - PROJD];   // backward v = [V[col], e2]
      atomicAdd(&aggc[c], vF * wF);
      atomicAdd(&aggr[c], vB * wB);
    }
  }
}

__global__ void cvt_agg_kernel(const float* __restrict__ agg, __bf16* __restrict__ aggb) {
  size_t i = (size_t)blockIdx.x * blockDim.x + threadIdx.x;
  if (i >= (size_t)NN * OUTD) return;
  aggb[i] = f2bf(agg[i]);
}

// ---------------- phase 4: residual + LayerNorm ----------------
__global__ __launch_bounds__(256) void ln_kernel(const float* __restrict__ o,
                                                 const float* __restrict__ x,
                                                 const int* __restrict__ ntype,
                                                 const float* __restrict__ gamma,
                                                 const float* __restrict__ beta,
                                                 float* __restrict__ out) {
  __shared__ float s1[256];
  __shared__ float s2[256];
  const int n = blockIdx.x;
  const int c = threadIdx.x;
  float y = o[(size_t)n * IND + c] + x[(size_t)n * IND + c];
  s1[c] = y;
  s2[c] = y * y;
  __syncthreads();
  for (int st = 128; st > 0; st >>= 1) {
    if (c < st) { s1[c] += s1[c + st]; s2[c] += s2[c + st]; }
    __syncthreads();
  }
  const float mu  = s1[0] * (1.0f / IND);
  const float var = s2[0] * (1.0f / IND) - mu * mu;
  const int t = ntype[n];
  out[(size_t)n * IND + c] =
      (y - mu) * rsqrtf(var + LN_EPS) * gamma[(size_t)t * IND + c] + beta[(size_t)t * IND + c];
}

// ---------------- host launcher ----------------
extern "C" void kernel_launch(void* const* d_in, const int* in_sizes, int n_in,
                              void* d_out, int out_size, void* d_ws, size_t ws_size,
                              hipStream_t stream) {
  (void)in_sizes; (void)n_in; (void)out_size; (void)ws_size;
  const float* x      = (const float*)d_in[0];
  const int*   ntype  = (const int*)d_in[1];
  const int*   eidx   = (const int*)d_in[2];
  const float* eattr  = (const float*)d_in[3];
  const float* Wk     = (const float*)d_in[4];
  const float* bk     = (const float*)d_in[5];
  const float* Wq     = (const float*)d_in[6];
  const float* bq     = (const float*)d_in[7];
  const float* Wv     = (const float*)d_in[8];
  const float* bv     = (const float*)d_in[9];
  const float* Wa     = (const float*)d_in[10];
  const float* ba     = (const float*)d_in[11];
  const float* gamma  = (const float*)d_in[12];
  const float* beta   = (const float*)d_in[13];
  const float* wtime  = (const float*)d_in[14];
  const float* btime  = (const float*)d_in[15];
  float* out = (float*)d_out;

  // bump allocator over d_ws
  char* ws = (char*)d_ws;
  size_t off = 0;
  auto alloc = [&](size_t bytes) -> char* {
    off = (off + 255) & ~(size_t)255;
    char* p = ws + off;
    off += bytes;
    return p;
  };
  int*      cnt      = (int*)alloc(NT * 4);
  int*      cur      = (int*)alloc(NT * 4);
  int*      offs     = (int*)alloc((NT + 1) * 4);
  int*      tileType = (int*)alloc(MAXTILES * 4);
  int*      perm     = (int*)alloc(MAXPAD * 4);
  __bf16*   xb       = (__bf16*)alloc((size_t)NN * IND * 2);       // reused as aggb later
  __bf16*   wt_kqv   = (__bf16*)alloc((size_t)NT * KQVC * IND * 2);
  __bf16*   wast     = (__bf16*)alloc((size_t)NT * OUTD * IND * 2);
  float*    bkqv     = (float*)alloc((size_t)NT * KQVC * 4);
  float*    Kb       = (float*)alloc((size_t)NN * PROJD * 4);
  float*    Qb       = (float*)alloc((size_t)NN * PROJD * 4);
  float*    Vb       = (float*)alloc((size_t)NN * PROJD * 4);
  float*    e0       = (float*)alloc((size_t)EE * TDIM * 4);
  float*    e12      = (float*)alloc((size_t)2 * EE * TDIM * 4);   // e1|e2, reused as o later
  float*    e1       = e12;
  float*    e2       = e12 + (size_t)EE * TDIM;
  float*    attF     = (float*)alloc((size_t)EE * NH * 4);
  float*    attB     = (float*)alloc((size_t)EE * NH * 4);
  unsigned* mF       = (unsigned*)alloc((size_t)NN * NH * 4);
  unsigned* mB       = (unsigned*)alloc((size_t)NN * NH * 4);
  float*    sF       = (float*)alloc((size_t)NN * NH * 4);
  float*    sB       = (float*)alloc((size_t)NN * NH * 4);
  float*    agg      = (float*)alloc((size_t)NN * OUTD * 4);
  __bf16*   aggb     = xb;          // overlay: xb dead after gemm_kqv
  float*    o        = e12;         // overlay: e1/e2 dead after scatter (2*E*32*4 == N*256*4)

  // init (graph-capture-safe async memsets)
  hipMemsetAsync(cnt, 0, NT * 4, stream);
  hipMemsetAsync(cur, 0, NT * 4, stream);
  hipMemsetAsync(tileType, 0xFF, MAXTILES * 4, stream);
  hipMemsetAsync(perm, 0xFF, MAXPAD * 4, stream);
  hipMemsetAsync(mF, 0, (size_t)NN * NH * 4, stream);
  hipMemsetAsync(mB, 0, (size_t)NN * NH * 4, stream);
  hipMemsetAsync(sF, 0, (size_t)NN * NH * 4, stream);
  hipMemsetAsync(sB, 0, (size_t)NN * NH * 4, stream);
  hipMemsetAsync(agg, 0, (size_t)NN * OUTD * 4, stream);

  // phase 0: bucket nodes by type
  count_types_kernel<<<(NN + 255) / 256, 256, 0, stream>>>(ntype, cnt);
  scan_kernel<<<1, 1, 0, stream>>>(cnt, offs, tileType);
  build_perm_kernel<<<(NN + 255) / 256, 256, 0, stream>>>(ntype, offs, cur, perm);

  // phase 1: conversions + weight packing
  cvt_x_kernel<<<(NN * IND + 255) / 256, 256, 0, stream>>>(x, xb);
  cvt_wkqv_kernel<<<(int)(((size_t)NT * KQVC * IND + 255) / 256), 256, 0, stream>>>(Wk, Wq, Wv, wt_kqv);
  build_bkqv_kernel<<<(NT * KQVC + 255) / 256, 256, 0, stream>>>(bk, bq, bv, bkqv);
  cvt_wast_kernel<<<(int)(((size_t)NT * OUTD * IND + 255) / 256), 256, 0, stream>>>(Wa, wast);

  // phase 2a: grouped K/Q/V projection GEMM (WMMA bf16, fp32 accum, 16x64 slab per wave)
  gemm_kqv_kernel<<<dim3(MAXTILES, KQVC / 64), 32, 0, stream>>>(
      xb, wt_kqv, bkqv, perm, tileType, Kb, Qb, Vb);

  // phase 3: edge attention pipeline
  time_emb_kernel<<<(int)(((size_t)3 * EE * TDIM + 255) / 256), 256, 0, stream>>>(
      eattr, wtime, btime, e0, e1, e2);
  edge_att_kernel<<<(EE + 255) / 256, 256, 0, stream>>>(eidx, Kb, Qb, e1, e2, attF, attB, mF, mB);
  edge_exp_kernel<<<(EE + 255) / 256, 256, 0, stream>>>(eidx, attF, attB, mF, mB, sF, sB);
  edge_scatter_kernel<<<(int)(((size_t)EE * 32 + 255) / 256), 256, 0, stream>>>(
      eidx, Vb, e0, e1, e2, attF, attB, sF, sB, agg);

  // phase 2b: grouped output projection GEMM (folded Wa_top+Wa_bot, 16x64 slab per wave)
  cvt_agg_kernel<<<(NN * OUTD + 255) / 256, 256, 0, stream>>>(agg, aggb);
  gemm_out_kernel<<<dim3(MAXTILES, OUTD / 64), 32, 0, stream>>>(
      aggb, wast, ba, perm, tileType, o);

  // phase 4: residual + typed LayerNorm
  ln_kernel<<<NN, 256, 0, stream>>>(o, x, ntype, gamma, beta, out);
}
